// CTANLayer_59219009077394
// MI455X (gfx1250) — matrix operand order repaired and checked by reading, not compiled, not run on hardware
//
#include <hip/hip_runtime.h>

#define B_SZ 4096
#define T_SZ 4
#define C_SZ 1024
#define H_SZ 8
#define DH_SZ 128
#define N_QKV (3 * C_SZ)
#define SCALE_F 0.08838834764831845f
#define LN_EPS_F 1e-5f

typedef __attribute__((ext_vector_type(16))) __bf16 v16bf;
typedef __attribute__((ext_vector_type(8)))  float  v8f;

#define BM 128
#define BN 128
#define BK 32
#define LDK 40   // padded row length (bf16 elems) -> 20 dwords, conflict-friendly

// Split fp32 into hi (truncated bf16) + lo (rounded bf16 of residual): x ~= hi + lo
__device__ __forceinline__ void splitf(float x, unsigned short& hi, unsigned short& lo) {
    unsigned u = __float_as_uint(x);
    unsigned short h = (unsigned short)(u >> 16);
    float r = x - __uint_as_float((unsigned)h << 16);
    unsigned ur = __float_as_uint(r);
    unsigned short l = (unsigned short)((ur + 0x7FFFu + ((ur >> 16) & 1u)) >> 16);
    hi = h; lo = l;
}

// Unified fragment load: both A and B tiles are stored in LDS so that each lane's
// 16 bf16 values are 8 contiguous dwords -> lowers to 2x ds_load_b128.
//   A rows store K in natural order:
//     lanes 0-15  read dwords 0..7  = K 0..15
//     lanes 16-31 read dwords 8..15 = K 16..31
//   B rows ([n][k], natural order):
//     lanes 0-15  read dwords 0..7  = K 0..15            (same order as A)
//     lanes 16-31 read dwords 8..15 = K 16..31
__device__ __forceinline__ v16bf frag(const unsigned short* Sb, int row, int lane) {
    const unsigned* p = (const unsigned*)(Sb + (row + (lane & 15)) * LDK) + ((lane >> 4) << 3);
    union { unsigned u[8]; v16bf v; } f;
#pragma unroll
    for (int j = 0; j < 8; ++j) f.u[j] = p[j];
    return f.v;
}

// Grouped GEMM: for task t = blockIdx.z:
//   Out[b, t, n] = sum_k A[b, t, k] * W[t, k, n] + bias[t, n]
// A layout [B, T, C] (row stride T*C, task offset t*C); W layout [T, C, N]; Out [B, T, N].
// fp32 emulated via bf16x3 split (hi*hi + hi*lo + lo*hi) on V_WMMA_F32_16X16X32_BF16.
// Register-level pipelining: next tile's global b128 loads issue before the WMMA
// block of the current tile and are converted/stored to LDS after the barrier.
__global__ __launch_bounds__(256) void gemm_bf16x3(
    const float* __restrict__ A, const float* __restrict__ W,
    const float* __restrict__ bias, float* __restrict__ Out, int N)
{
    __shared__ unsigned short As_hi[BM * LDK];
    __shared__ unsigned short As_lo[BM * LDK];
    __shared__ unsigned short Bs_hi[BN * LDK];
    __shared__ unsigned short Bs_lo[BN * LDK];

    const int t    = blockIdx.z;
    const int n0   = blockIdx.x * BN;
    const int m0   = blockIdx.y * BM;
    const int tid  = threadIdx.x;
    const int lane = tid & 31;
    const int wave = tid >> 5;
    const int wm   = wave >> 2;   // 0..1  (64-row band)
    const int wn   = wave & 3;    // 0..3  (32-col band)

    const size_t ldA = (size_t)T_SZ * C_SZ;
    const float* Ab    = A + (size_t)t * C_SZ;
    const float* Wb    = W + (size_t)t * C_SZ * N;
    const float* biasb = bias + (size_t)t * N;
    float*       Ob    = Out + (size_t)t * N;
    const size_t ldO   = (size_t)T_SZ * N;

    const v8f vzero = {0.f, 0.f, 0.f, 0.f, 0.f, 0.f, 0.f, 0.f};
    v8f acc[4][2];
#pragma unroll
    for (int mt = 0; mt < 4; ++mt)
#pragma unroll
        for (int nt = 0; nt < 2; ++nt) acc[mt][nt] = vzero;

    float4 ar[4], br[4];

    // ---- prologue: fetch K-tile 0 into registers, stage to LDS ----
#pragma unroll
    for (int i = 0; i < 4; ++i) {
        int idx = tid + i * 256;
        int r = idx >> 3, c4 = idx & 7;
        ar[i] = *(const float4*)(Ab + (size_t)(m0 + r) * ldA + c4 * 4);
    }
#pragma unroll
    for (int i = 0; i < 4; ++i) {
        int idx = tid + i * 256;
        int kr = idx >> 5, c4 = idx & 31;
        br[i] = *(const float4*)(Wb + (size_t)kr * N + n0 + c4 * 4);
    }

    for (int k0 = 0; k0 < C_SZ; k0 += BK) {
        // ---- stage registers -> LDS (split hi/lo bf16) ----
#pragma unroll
        for (int i = 0; i < 4; ++i) {
            int idx = tid + i * 256;
            int r = idx >> 3, c4 = idx & 7;
            int g = c4 >> 1;                                   // K-group of 8
            int posBase = c4 * 4 + (g - g);
            const float av[4] = {ar[i].x, ar[i].y, ar[i].z, ar[i].w};
            unsigned short h[4], l[4];
#pragma unroll
            for (int e = 0; e < 4; ++e) splitf(av[e], h[e], l[e]);
            unsigned* dh = (unsigned*)(As_hi + r * LDK + posBase);
            unsigned* dl = (unsigned*)(As_lo + r * LDK + posBase);
            dh[0] = (unsigned)h[0] | ((unsigned)h[1] << 16);
            dh[1] = (unsigned)h[2] | ((unsigned)h[3] << 16);
            dl[0] = (unsigned)l[0] | ((unsigned)l[1] << 16);
            dl[1] = (unsigned)l[2] | ((unsigned)l[3] << 16);
        }
#pragma unroll
        for (int i = 0; i < 4; ++i) {
            int idx = tid + i * 256;
            int kr = idx >> 5, c4 = idx & 31;
            const float bv[4] = {br[i].x, br[i].y, br[i].z, br[i].w};
#pragma unroll
            for (int e = 0; e < 4; ++e) {
                int n = c4 * 4 + e;
                unsigned short hh, ll; splitf(bv[e], hh, ll);
                Bs_hi[n * LDK + kr] = hh;
                Bs_lo[n * LDK + kr] = ll;
            }
        }
        __syncthreads();

        // ---- pipeline: issue next K-tile's global loads now ----
        const bool more = (k0 + BK) < C_SZ;
        if (more) {
            const int k1 = k0 + BK;
#pragma unroll
            for (int i = 0; i < 4; ++i) {
                int idx = tid + i * 256;
                int r = idx >> 3, c4 = idx & 7;
                ar[i] = *(const float4*)(Ab + (size_t)(m0 + r) * ldA + k1 + c4 * 4);
            }
#pragma unroll
            for (int i = 0; i < 4; ++i) {
                int idx = tid + i * 256;
                int kr = idx >> 5, c4 = idx & 31;
                br[i] = *(const float4*)(Wb + (size_t)(k1 + kr) * N + n0 + c4 * 4);
            }
        }

        // ---- compute: 8 tiles x 3 WMMA (bf16x3) from LDS ----
        v16bf ah[4], al[4], bh[2], bl[2];
#pragma unroll
        for (int mt = 0; mt < 4; ++mt) {
            ah[mt] = frag(As_hi, wm * 64 + mt * 16, lane);
            al[mt] = frag(As_lo, wm * 64 + mt * 16, lane);
        }
#pragma unroll
        for (int nt = 0; nt < 2; ++nt) {
            bh[nt] = frag(Bs_hi, wn * 32 + nt * 16, lane);
            bl[nt] = frag(Bs_lo, wn * 32 + nt * 16, lane);
        }
#pragma unroll
        for (int mt = 0; mt < 4; ++mt)
#pragma unroll
            for (int nt = 0; nt < 2; ++nt) {
                acc[mt][nt] = __builtin_amdgcn_wmma_f32_16x16x32_bf16(
                    false, ah[mt], false, bh[nt], (short)0, acc[mt][nt], false, false);
                acc[mt][nt] = __builtin_amdgcn_wmma_f32_16x16x32_bf16(
                    false, ah[mt], false, bl[nt], (short)0, acc[mt][nt], false, false);
                acc[mt][nt] = __builtin_amdgcn_wmma_f32_16x16x32_bf16(
                    false, al[mt], false, bh[nt], (short)0, acc[mt][nt], false, false);
            }
        __syncthreads();
    }

    // C/D layout: VGPR r -> M = r (lanes 0-15) / r+8 (lanes 16-31), N = lane&15
#pragma unroll
    for (int mt = 0; mt < 4; ++mt)
#pragma unroll
        for (int nt = 0; nt < 2; ++nt) {
            int col   = n0 + wn * 32 + nt * 16 + (lane & 15);
            float bv  = biasb[col];
            int rbase = m0 + wm * 64 + mt * 16 + ((lane >> 4) << 3);
#pragma unroll
            for (int r = 0; r < 8; ++r)
                Ob[(size_t)(rbase + r) * ldO + col] = acc[mt][nt][r] + bv;
        }
}

// Cross-task attention: one wave per (b, h); T=4 so the whole score matrix fits in regs.
__global__ __launch_bounds__(256) void attn_kernel(const float* __restrict__ qkv,
                                                   float* __restrict__ ctx)
{
    const int b    = blockIdx.x;
    const int h    = threadIdx.x >> 5;   // 8 waves = 8 heads
    const int lane = threadIdx.x & 31;   // 32 lanes * 4 floats = DH=128
    const float* base = qkv + (size_t)b * T_SZ * N_QKV + h * DH_SZ + lane * 4;

    float4 q[T_SZ], k[T_SZ], v[T_SZ];
#pragma unroll
    for (int t = 0; t < T_SZ; ++t) {
        const float* row = base + (size_t)t * N_QKV;
        q[t] = *(const float4*)(row);
        k[t] = *(const float4*)(row + C_SZ);
        v[t] = *(const float4*)(row + 2 * C_SZ);
    }
    float s[T_SZ][T_SZ];
#pragma unroll
    for (int qt = 0; qt < T_SZ; ++qt)
#pragma unroll
        for (int kt = 0; kt < T_SZ; ++kt)
            s[qt][kt] = q[qt].x * k[kt].x + q[qt].y * k[kt].y +
                        q[qt].z * k[kt].z + q[qt].w * k[kt].w;
#pragma unroll
    for (int off = 16; off; off >>= 1)
#pragma unroll
        for (int qt = 0; qt < T_SZ; ++qt)
#pragma unroll
            for (int kt = 0; kt < T_SZ; ++kt)
                s[qt][kt] += __shfl_xor(s[qt][kt], off, 32);

#pragma unroll
    for (int qt = 0; qt < T_SZ; ++qt) {
        float sc[T_SZ];
#pragma unroll
        for (int kt = 0; kt < T_SZ; ++kt) sc[kt] = s[qt][kt] * SCALE_F;
        float mx = sc[0];
#pragma unroll
        for (int kt = 1; kt < T_SZ; ++kt) mx = fmaxf(mx, sc[kt]);
        float p[T_SZ], sum = 0.f;
#pragma unroll
        for (int kt = 0; kt < T_SZ; ++kt) { p[kt] = __expf(sc[kt] - mx); sum += p[kt]; }
        const float inv = 1.0f / sum;
        float4 o = {0.f, 0.f, 0.f, 0.f};
#pragma unroll
        for (int kt = 0; kt < T_SZ; ++kt) {
            float w = p[kt] * inv;
            o.x += w * v[kt].x; o.y += w * v[kt].y; o.z += w * v[kt].z; o.w += w * v[kt].w;
        }
        *(float4*)(ctx + ((size_t)b * T_SZ + qt) * C_SZ + h * DH_SZ + lane * 4) = o;
    }
}

// Residual + LayerNorm over the channel dim; one 256-thread block per (b,t) row.
__global__ __launch_bounds__(256) void ln_kernel(const float* __restrict__ y,
                                                 const float* __restrict__ feats,
                                                 const float* __restrict__ gamma,
                                                 const float* __restrict__ beta,
                                                 float* __restrict__ out)
{
    const int row  = blockIdx.x;
    const size_t base = (size_t)row * C_SZ;
    const int tid  = threadIdx.x;
    const int lane = tid & 31, wave = tid >> 5;

    float x[4], s = 0.f, ss = 0.f;
#pragma unroll
    for (int i = 0; i < 4; ++i) {
        int c = tid + i * 256;
        float xv = y[base + c] + feats[base + c];
        x[i] = xv; s += xv; ss += xv * xv;
    }
#pragma unroll
    for (int off = 16; off; off >>= 1) {
        s  += __shfl_xor(s,  off, 32);
        ss += __shfl_xor(ss, off, 32);
    }
    __shared__ float sred[8], ssred[8];
    if (lane == 0) { sred[wave] = s; ssred[wave] = ss; }
    __syncthreads();
    float tot = 0.f, tot2 = 0.f;
#pragma unroll
    for (int w = 0; w < 8; ++w) { tot += sred[w]; tot2 += ssred[w]; }
    const float mu  = tot * (1.0f / C_SZ);
    const float var = tot2 * (1.0f / C_SZ) - mu * mu;
    const float rs  = rsqrtf(var + LN_EPS_F);
#pragma unroll
    for (int i = 0; i < 4; ++i) {
        int c = tid + i * 256;
        out[base + c] = (x[i] - mu) * rs * gamma[c] + beta[c];
    }
}

extern "C" void kernel_launch(void* const* d_in, const int* in_sizes, int n_in,
                              void* d_out, int out_size, void* d_ws, size_t ws_size,
                              hipStream_t stream) {
    (void)in_sizes; (void)n_in; (void)out_size; (void)ws_size;
    const float* feats = (const float*)d_in[0];
    const float* Wqkv  = (const float*)d_in[1];
    const float* bqkv  = (const float*)d_in[2];
    const float* Wproj = (const float*)d_in[3];
    const float* bproj = (const float*)d_in[4];
    const float* gamma = (const float*)d_in[5];
    const float* beta  = (const float*)d_in[6];
    float* out = (float*)d_out;

    float* qkv = (float*)d_ws;                                 // [B,T,3C] = 192 MB
    float* ctx = qkv + (size_t)B_SZ * T_SZ * N_QKV;            // [B,T,C]  =  64 MB
    float* y   = qkv;                                          // proj out reuses qkv

    gemm_bf16x3<<<dim3(N_QKV / BN, B_SZ / BM, T_SZ), 256, 0, stream>>>(
        feats, Wqkv, bqkv, qkv, N_QKV);
    attn_kernel<<<B_SZ, 256, 0, stream>>>(qkv, ctx);
    gemm_bf16x3<<<dim3(C_SZ / BN, B_SZ / BM, T_SZ), 256, 0, stream>>>(
        ctx, Wproj, bproj, y, C_SZ);
    ln_kernel<<<B_SZ * T_SZ, 256, 0, stream>>>(y, feats, gamma, beta, out);
}
